// MyBertSelfAttention_15453292331091
// MI455X (gfx1250) — matrix operand order repaired and checked by reading, not compiled, not run on hardware
//
#include <hip/hip_runtime.h>

// ---------------------------------------------------------------------------
// Problem constants (from reference): B=8, L=1024, D=1024, H=16, HD=64
// ---------------------------------------------------------------------------
#define BATCH 8
#define SEQ   1024
#define DIM   1024
#define HEADS 16
#define HDIM  64

typedef _Float16 f16;
typedef unsigned int u32;
typedef __attribute__((ext_vector_type(16))) _Float16 v16h;
typedef __attribute__((ext_vector_type(8)))  float    v8f;
typedef __attribute__((ext_vector_type(4)))  u32      v4u;
typedef __attribute__((ext_vector_type(8)))  int      v8i;
typedef __attribute__((ext_vector_type(4)))  int      v4i;

// ---------------------------------------------------------------------------
// CDNA5 WMMA fragment index helpers (wave32, V_WMMA_F32_16X16X32_F16)
//   A (16x32 f16): lane m=lane&15, hi=lane>>4; element e -> K = hi*16 + e
//   B (32x16 f16): lane n=lane&15, element e -> K = hi*16 + e
//   C/D (16x16 f32): VGPR r -> row = r + hi*8, col = lane&15
// ---------------------------------------------------------------------------
__device__ __forceinline__ int a_frag_k(int e, int hi) {
  return hi * 16 + e;
}

// ---------------------------------------------------------------------------
// Kernel 1: projection GEMM  Y = X @ W + bias  (f32 in, f16 split-head out)
// Block: 128 threads (4 waves), tile 64(M) x 64(N), K step 32.
// Software-pipelined: next K-tile's global loads overlap current tile's WMMAs.
// ---------------------------------------------------------------------------
__global__ __launch_bounds__(128)
void proj_gemm_f16_kernel(const float* __restrict__ X,
                          const float* __restrict__ W,
                          const float* __restrict__ bias,
                          f16* __restrict__ Yh) {
  __shared__ f16 As[64][32 + 4];   // row stride 72B (8B aligned)
  __shared__ f16 Bs[32][64 + 4];   // row stride 136B (8B aligned)

  const int n0   = blockIdx.x * 64;
  const int r0   = blockIdx.y * 64;
  const int tid  = threadIdx.x;
  const int wave = tid >> 5;
  const int lane = tid & 31;
  const int m16  = lane & 15;
  const int hi   = lane >> 4;

  v8f acc[4];
  const v8f vzero = {};
#pragma unroll
  for (int t = 0; t < 4; ++t) acc[t] = vzero;

  // Per-thread register staging: 4 float4 of A-tile, 4 float4 of W-tile.
  float4 ra[4], rw[4];

  auto issue_tile_loads = [&](int k0) {
#pragma unroll
    for (int i = 0; i < 4; ++i) {       // A tile: 64 x 32, 8 float4 per row
      const int c   = tid + 128 * i;
      const int row = c >> 3;
      const int col = (c & 7) * 4;
      ra[i] = *(const float4*)&X[(size_t)(r0 + row) * DIM + k0 + col];
    }
#pragma unroll
    for (int i = 0; i < 4; ++i) {       // W tile: 32 x 64, 16 float4 per row
      const int c   = tid + 128 * i;
      const int row = c >> 4;
      const int col = (c & 15) * 4;
      rw[i] = *(const float4*)&W[(size_t)(k0 + row) * DIM + n0 + col];
    }
  };

  issue_tile_loads(0);                  // prologue

  for (int k0 = 0; k0 < DIM; k0 += 32) {
    // ---- convert staged registers -> LDS f16
#pragma unroll
    for (int i = 0; i < 4; ++i) {
      const int c   = tid + 128 * i;
      const int row = c >> 3;
      const int col = (c & 7) * 4;
      As[row][col + 0] = (f16)ra[i].x;
      As[row][col + 1] = (f16)ra[i].y;
      As[row][col + 2] = (f16)ra[i].z;
      As[row][col + 3] = (f16)ra[i].w;
    }
#pragma unroll
    for (int i = 0; i < 4; ++i) {
      const int c   = tid + 128 * i;
      const int row = c >> 4;
      const int col = (c & 15) * 4;
      Bs[row][col + 0] = (f16)rw[i].x;
      Bs[row][col + 1] = (f16)rw[i].y;
      Bs[row][col + 2] = (f16)rw[i].z;
      Bs[row][col + 3] = (f16)rw[i].w;
    }
    __syncthreads();

    // ---- prefetch next K-tile (overlaps with fragment builds + WMMAs)
    if (k0 + 32 < DIM) issue_tile_loads(k0 + 32);

    // ---- A fragment for this wave's 16-row slice
    v16h a;
    const int arow = wave * 16 + m16;
#pragma unroll
    for (int e = 0; e < 16; ++e) a[e] = As[arow][a_frag_k(e, hi)];

    // ---- 4 N-subtiles of 16, reusing the A fragment
#pragma unroll
    for (int t = 0; t < 4; ++t) {
      v16h b;
#pragma unroll
      for (int e = 0; e < 16; ++e) b[e] = Bs[hi * 16 + e][t * 16 + m16];
      acc[t] = __builtin_amdgcn_wmma_f32_16x16x32_f16(
          false, a, false, b, (short)0, acc[t], false, false);
    }
    __syncthreads();
  }

  // ---- bias + store split-head f16
#pragma unroll
  for (int t = 0; t < 4; ++t) {
#pragma unroll
    for (int r = 0; r < 8; ++r) {
      const int rowG = r0 + wave * 16 + r + hi * 8;
      const int colG = n0 + t * 16 + m16;
      const int b  = rowG >> 10;
      const int l  = rowG & (SEQ - 1);
      const int h  = colG >> 6;
      const int hd = colG & (HDIM - 1);
      const float v = acc[t][r] + bias[colG];
      Yh[((((size_t)b * HEADS + h) * SEQ) + l) * HDIM + hd] = (f16)v;
    }
  }
}

// ---------------------------------------------------------------------------
// Kernel 2: attention. One wave per (b, h, 16-query tile).
//   scores = (Q Kᵀ + Q2 K2ᵀ) * scale + mask ; softmax ; ctx = P V
// V tiles (32 keys x 64 hd x f16 = 4KB) are staged into ping-pong LDS buffers
// by the Tensor Data Mover (tensor_load_to_lds), double-buffered with
// s_wait_tensorcnt.
// ---------------------------------------------------------------------------
__global__ __launch_bounds__(32)
void attn_f16_wmma_kernel(const f16* __restrict__ Qh,
                          const f16* __restrict__ Kh,
                          const f16* __restrict__ Vh,
                          const f16* __restrict__ Q2h,
                          const f16* __restrict__ K2h,
                          const float* __restrict__ mask,
                          float* __restrict__ out) {
  __shared__ f16 sc[16][SEQ + 8];      // 33 KB score/prob tile
  __shared__ f16 Vs[2][32][HDIM];      // 8 KB ping-pong V tiles (TDM dest)

  const int q0   = blockIdx.x * 16;
  const int h    = blockIdx.y;
  const int b    = blockIdx.z;
  const int lane = threadIdx.x;
  const int m16  = lane & 15;
  const int hi   = lane >> 4;

  const size_t headOff = (((size_t)b * HEADS + h) * SEQ) * HDIM;
  const float  scale   = 0.125f;       // 1/sqrt(64)
  const v8f    vzero   = {};

  // ---- TDM issue helper: 2D tile (tile_dim0=HDIM elems, tile_dim1=32 rows)
  // from V[key][hd] (row stride HDIM) into Vs[buf]. Descriptor per ISA §8.
  auto issue_v_tile = [&](int buf, int k0) {
    const unsigned long long gaddr =
        (unsigned long long)(size_t)(Vh + headOff + (size_t)k0 * HDIM);
    const u32 ldsOff = (u32)(size_t)(&Vs[buf][0][0]);
    // group0: [1:0]=count=1 | lds_addr | global_addr[56:0] | type=2@[127:126]
    v4u g0 = {1u, ldsOff, (u32)gaddr, (u32)(gaddr >> 32) | (2u << 30)};
    // group1: data_size=2B(code1)@[17:16]; tensor_dim0=HDIM@[79:48];
    // tensor_dim1=SEQ@[111:80]; tile_dim0=HDIM@[127:112]; tile_dim1=32@[143:128];
    // tensor_dim0_stride=HDIM@[207:160]
    v8i g1 = {(int)(1u << 16),
              (int)((u32)HDIM << 16),
              (int)((u32)SEQ << 16),
              (int)((u32)HDIM << 16),
              32,
              HDIM,
              0,
              0};
    v4i gz4 = {0, 0, 0, 0};
    v8i gz8 = {0, 0, 0, 0, 0, 0, 0, 0};
    __builtin_amdgcn_tensor_load_to_lds(g0, g1, gz4, gz4, gz8, 0);
  };

  // Kick off the first V tile now: the TDM runs in the background during the
  // whole scores + softmax phase.
  issue_v_tile(0, 0);

  // ---- Q / Q2 fragments (A operand, k-dim = HD = 64 -> 2 K-steps)
  v16h aq[2], aq2[2];
#pragma unroll
  for (int ks = 0; ks < 2; ++ks) {
    const size_t rbase = headOff + (size_t)(q0 + m16) * HDIM + ks * 32;
#pragma unroll
    for (int e = 0; e < 16; ++e) {
      const int kl = a_frag_k(e, hi);
      aq[ks][e]  = Qh[rbase + kl];
      aq2[ks][e] = Q2h[rbase + kl];
    }
  }

  const float* maskRow = mask + (size_t)b * SEQ;

  // ---- scores: loop over 16-key tiles
  for (int n0 = 0; n0 < SEQ; n0 += 16) {
    v8f accs = vzero;
#pragma unroll
    for (int ks = 0; ks < 2; ++ks) {
      // B operand = Kᵀ: per lane 16 contiguous f16 (32B aligned) -> vector load
      const v16h bk =
          *(const v16h*)(Kh + headOff + (size_t)(n0 + m16) * HDIM + ks * 32 + hi * 16);
      const v16h bk2 =
          *(const v16h*)(K2h + headOff + (size_t)(n0 + m16) * HDIM + ks * 32 + hi * 16);
      accs = __builtin_amdgcn_wmma_f32_16x16x32_f16(
          false, aq[ks], false, bk, (short)0, accs, false, false);
      accs = __builtin_amdgcn_wmma_f32_16x16x32_f16(
          false, aq2[ks], false, bk2, (short)0, accs, false, false);
    }
    const float mval = maskRow[n0 + m16];
#pragma unroll
    for (int r = 0; r < 8; ++r)
      sc[r + hi * 8][n0 + m16] = (f16)(accs[r] * scale + mval);
  }
  __syncthreads();

  // ---- softmax: lanes 0..15 each own one query row
  if (lane < 16) {
    float mx = -1e30f;
    for (int c = 0; c < SEQ; ++c) mx = fmaxf(mx, (float)sc[lane][c]);
    float sum = 0.f;
    for (int c = 0; c < SEQ; ++c) {
      const float e = __expf((float)sc[lane][c] - mx);
      sc[lane][c] = (f16)e;
      sum += e;
    }
    const float inv = 1.f / sum;
    for (int c = 0; c < SEQ; ++c) sc[lane][c] = (f16)((float)sc[lane][c] * inv);
  }
  __syncthreads();

  // ---- ctx = probs @ V : K-dim = 1024 keys (32 tiles), 4 N-subtiles of HD.
  // Tile i lives in Vs[i&1]; tile i+1 issued while tile i is consumed.
  v8f acc[4];
#pragma unroll
  for (int t = 0; t < 4; ++t) acc[t] = vzero;

  for (int k0 = 0; k0 < SEQ; k0 += 32) {
    const int buf = (k0 >> 5) & 1;
    if (k0 + 32 < SEQ) {
      // guard ping-pong buffer reuse against in-flight DS reads, then
      // prefetch the next tile and wait only for the current one (TDM ops
      // from one wave complete in order).
      asm volatile("s_wait_dscnt 0x0" ::: "memory");
      issue_v_tile(buf ^ 1, k0 + 32);
      __builtin_amdgcn_s_wait_tensorcnt(1);
    } else {
      __builtin_amdgcn_s_wait_tensorcnt(0);
    }
    asm volatile("" ::: "memory");     // keep LDS reads below the wait

    v16h a;
#pragma unroll
    for (int e = 0; e < 16; ++e) a[e] = sc[m16][k0 + a_frag_k(e, hi)];
#pragma unroll
    for (int t = 0; t < 4; ++t) {
      v16h bv;
#pragma unroll
      for (int e = 0; e < 16; ++e) bv[e] = Vs[buf][hi * 16 + e][t * 16 + m16];
      acc[t] = __builtin_amdgcn_wmma_f32_16x16x32_f16(
          false, a, false, bv, (short)0, acc[t], false, false);
    }
  }

  // ---- store ctx -> out[b][l][h*HD + hd]  (f32)
#pragma unroll
  for (int t = 0; t < 4; ++t) {
#pragma unroll
    for (int r = 0; r < 8; ++r) {
      const int row = q0 + r + hi * 8;
      const int col = h * HDIM + t * 16 + m16;
      out[((size_t)b * SEQ + row) * DIM + col] = acc[t][r];
    }
  }
}

// ---------------------------------------------------------------------------
// Launch: 5 projection GEMMs into f16 workspace, then fused attention.
// Input order: hidden(0) mask(1) source(2) Wq(3) bq(4) Wk(5) bk(6)
//              Wv(7) bv(8) Wq2(9) bq2(10) Wk2(11) bk2(12)
// ---------------------------------------------------------------------------
extern "C" void kernel_launch(void* const* d_in, const int* in_sizes, int n_in,
                              void* d_out, int out_size, void* d_ws, size_t ws_size,
                              hipStream_t stream) {
  const float* hidden = (const float*)d_in[0];
  const float* mask   = (const float*)d_in[1];
  const float* source = (const float*)d_in[2];
  const float* Wq  = (const float*)d_in[3];
  const float* bq  = (const float*)d_in[4];
  const float* Wk  = (const float*)d_in[5];
  const float* bk  = (const float*)d_in[6];
  const float* Wv  = (const float*)d_in[7];
  const float* bv  = (const float*)d_in[8];
  const float* Wq2 = (const float*)d_in[9];
  const float* bq2 = (const float*)d_in[10];
  const float* Wk2 = (const float*)d_in[11];
  const float* bk2 = (const float*)d_in[12];
  float* out = (float*)d_out;

  const size_t nElem = (size_t)BATCH * HEADS * SEQ * HDIM;  // 8,388,608
  f16* Qh  = (f16*)d_ws;
  f16* Kh  = Qh + nElem;
  f16* Vh  = Kh + nElem;
  f16* Q2h = Vh + nElem;
  f16* K2h = Q2h + nElem;

  dim3 gProj(DIM / 64, (BATCH * SEQ) / 64);   // (16, 128)
  proj_gemm_f16_kernel<<<gProj, 128, 0, stream>>>(hidden, Wq,  bq,  Qh);
  proj_gemm_f16_kernel<<<gProj, 128, 0, stream>>>(hidden, Wk,  bk,  Kh);
  proj_gemm_f16_kernel<<<gProj, 128, 0, stream>>>(hidden, Wv,  bv,  Vh);
  proj_gemm_f16_kernel<<<gProj, 128, 0, stream>>>(hidden, Wq2, bq2, Q2h);
  proj_gemm_f16_kernel<<<gProj, 128, 0, stream>>>(source, Wk2, bk2, K2h);

  dim3 gAttn(SEQ / 16, HEADS, BATCH);          // (64, 16, 8)
  attn_f16_wmma_kernel<<<gAttn, 32, 0, stream>>>(Qh, Kh, Vh, Q2h, K2h, mask, out);
}
